// Router_19155554140173
// MI455X (gfx1250) — matrix operand order repaired and checked by reading, not compiled, not run on hardware
//
#include <hip/hip_runtime.h>

// MoE router for MI455X (gfx1250, wave32, WMMA).
// tokens [32768 x 2048] f32, jitter same, w [2048 x 64] f32, b [64] f32.
// out = softmax(x@w + b) * top2_mask, [32768 x 64] f32.

typedef __attribute__((ext_vector_type(16))) _Float16 v16h;
typedef __attribute__((ext_vector_type(8)))  float    v8f;

#define H_DIM 2048
#define E_DIM 64

// ---------------------------------------------------------------------------
// Pass 1: w [H][E] f32 (row-major, E fastest) -> wT [E][H] f16 in workspace.
// Makes per-lane B-fragment loads contiguous (32B aligned) in pass 2.
// ---------------------------------------------------------------------------
__global__ __launch_bounds__(256) void wT_convert_kernel(
    const float* __restrict__ W, _Float16* __restrict__ WT)
{
    int idx = blockIdx.x * 256 + threadIdx.x;   // coalesced read of w
    if (idx < H_DIM * E_DIM) {
        int k = idx >> 6;   // / E_DIM
        int e = idx & 63;   // % E_DIM
        WT[e * H_DIM + k] = (_Float16)W[idx];
    }
}

// ---------------------------------------------------------------------------
// Pass 2: fused jitter-mul + GEMM (f16 WMMA, f32 accum) + bias + softmax +
// top-2 mask. One block = 256 threads = 8 waves; wave handles 16 tokens x 64
// experts (4 WMMA C tiles). Block handles 128 tokens.
// ---------------------------------------------------------------------------
__global__ __launch_bounds__(256) void router_kernel(
    const float* __restrict__ X, const float* __restrict__ J,
    const _Float16* __restrict__ WT, const float* __restrict__ bias,
    float* __restrict__ out)
{
    const int lane = threadIdx.x & 31;
    const int wave = threadIdx.x >> 5;
    const int l16  = lane & 15;
    const int hi   = lane >> 4;          // 0: lanes 0-15, 1: lanes 16-31

    // A-fragment row for this lane (same row for lane and lane+16).
    const long rowA = (long)blockIdx.x * 128 + wave * 16 + l16;
    const float* xrow = X + rowA * H_DIM;
    const float* jrow = J + rowA * H_DIM;

    // B-fragment base pointers: lane holds column e = l16 + 16*t of w,
    // 16 consecutive K halves starting at hi*16 within each 32-K chunk.
    const _Float16* wt0 = WT + (long)(l16 +  0) * H_DIM + hi * 16;
    const _Float16* wt1 = WT + (long)(l16 + 16) * H_DIM + hi * 16;
    const _Float16* wt2 = WT + (long)(l16 + 32) * H_DIM + hi * 16;
    const _Float16* wt3 = WT + (long)(l16 + 48) * H_DIM + hi * 16;

    // A k order used here (contiguous, same as B): lanes 0-15 hold K {0..15}; lanes 16-31 hold
    // K {16..31} of each 32-wide chunk.
    const int ka = hi * 16;

    v8f c0 = {}, c1 = {}, c2 = {}, c3 = {};

    #pragma unroll 4
    for (int k0 = 0; k0 < H_DIM; k0 += 32) {
        const float4* xp = (const float4*)(xrow + k0 + ka);
        const float4* jp = (const float4*)(jrow + k0 + ka);
        float4 x0 = xp[0], x1 = xp[1], x2 = xp[2], x3 = xp[3];
        float4 j0 = jp[0], j1 = jp[1], j2 = jp[2], j3 = jp[3];

        v16h a;
        a[0]  = (_Float16)(x0.x * j0.x);
        a[1]  = (_Float16)(x0.y * j0.y);
        a[2]  = (_Float16)(x0.z * j0.z);
        a[3]  = (_Float16)(x0.w * j0.w);
        a[4]  = (_Float16)(x1.x * j1.x);
        a[5]  = (_Float16)(x1.y * j1.y);
        a[6]  = (_Float16)(x1.z * j1.z);
        a[7]  = (_Float16)(x1.w * j1.w);
        a[8]  = (_Float16)(x2.x * j2.x);
        a[9]  = (_Float16)(x2.y * j2.y);
        a[10] = (_Float16)(x2.z * j2.z);
        a[11] = (_Float16)(x2.w * j2.w);
        a[12] = (_Float16)(x3.x * j3.x);
        a[13] = (_Float16)(x3.y * j3.y);
        a[14] = (_Float16)(x3.z * j3.z);
        a[15] = (_Float16)(x3.w * j3.w);

        v16h b0 = *(const v16h*)(wt0 + k0);   // 32B contiguous, L2-hot
        v16h b1 = *(const v16h*)(wt1 + k0);
        v16h b2 = *(const v16h*)(wt2 + k0);
        v16h b3 = *(const v16h*)(wt3 + k0);

        // 4 independent WMMAs (shared A) -> good XDL pipelining.
        c0 = __builtin_amdgcn_wmma_f32_16x16x32_f16(false, a, false, b0,
                                                    (short)0, c0, false, false);
        c1 = __builtin_amdgcn_wmma_f32_16x16x32_f16(false, a, false, b1,
                                                    (short)0, c1, false, false);
        c2 = __builtin_amdgcn_wmma_f32_16x16x32_f16(false, a, false, b2,
                                                    (short)0, c2, false, false);
        c3 = __builtin_amdgcn_wmma_f32_16x16x32_f16(false, a, false, b3,
                                                    (short)0, c3, false, false);
    }

    // Per-lane expert column is constant per tile: n = l16 + 16*t.
    const float bs0 = bias[l16 +  0];
    const float bs1 = bias[l16 + 16];
    const float bs2 = bias[l16 + 32];
    const float bs3 = bias[l16 + 48];

    // C layout: VGPR r holds row M = r + 8*hi, col N = l16 (+16 per tile).
    const long rowOutBase = (long)blockIdx.x * 128 + wave * 16 + hi * 8;

    #pragma unroll
    for (int r = 0; r < 8; ++r) {
        float v0 = c0[r] + bs0;
        float v1 = c1[r] + bs1;
        float v2 = c2[r] + bs2;
        float v3 = c3[r] + bs3;

        // per-lane top-2 of the 4 local logits
        float a1 = fmaxf(v0, v1), a2 = fminf(v0, v1);
        float b1 = fmaxf(v2, v3), b2 = fminf(v2, v3);
        float m1 = fmaxf(a1, b1);
        float m2 = fmaxf(fminf(a1, b1), fmaxf(a2, b2));

        // merge top-2 across the 16-lane half-wave (xor<=8 stays in-half)
        #pragma unroll
        for (int off = 8; off >= 1; off >>= 1) {
            float o1 = __shfl_xor(m1, off, 32);
            float o2 = __shfl_xor(m2, off, 32);
            m2 = fmaxf(fminf(m1, o1), fmaxf(m2, o2));
            m1 = fmaxf(m1, o1);
        }
        // m1 == row max, m2 == 2nd largest

        float e0 = __expf(v0 - m1);
        float e1 = __expf(v1 - m1);
        float e2 = __expf(v2 - m1);
        float e3 = __expf(v3 - m1);
        float s = e0 + e1 + e2 + e3;
        #pragma unroll
        for (int off = 8; off >= 1; off >>= 1)
            s += __shfl_xor(s, off, 32);
        float inv = 1.0f / s;

        float* orow = out + (rowOutBase + r) * E_DIM + l16;
        orow[0]  = (v0 >= m2) ? e0 * inv : 0.0f;
        orow[16] = (v1 >= m2) ? e1 * inv : 0.0f;
        orow[32] = (v2 >= m2) ? e2 * inv : 0.0f;
        orow[48] = (v3 >= m2) ? e3 * inv : 0.0f;
    }
}

extern "C" void kernel_launch(void* const* d_in, const int* in_sizes, int n_in,
                              void* d_out, int out_size, void* d_ws, size_t ws_size,
                              hipStream_t stream) {
    const float* X    = (const float*)d_in[0];   // token_inputs [B,S,H]
    const float* Jit  = (const float*)d_in[1];   // jitter       [B,S,H]
    const float* W    = (const float*)d_in[2];   // w            [H,E]
    const float* Bias = (const float*)d_in[3];   // b            [E]
    float* out = (float*)d_out;

    _Float16* WT = (_Float16*)d_ws;              // 64*2048*2 = 256 KB

    const long tokens = (long)in_sizes[0] / H_DIM;   // 32768

    wT_convert_kernel<<<(H_DIM * E_DIM + 255) / 256, 256, 0, stream>>>(W, WT);
    router_kernel<<<(int)(tokens / 128), 256, 0, stream>>>(X, Jit, WT, Bias, out);
}
